// MPCTCLSTMSELF_60825326846472
// MI455X (gfx1250) — matrix operand order repaired and checked by reading, not compiled, not run on hardware
//
#include <hip/hip_runtime.h>
#include <hip/hip_bf16.h>
#include <math.h>

#define Tt 512
#define Bb 32
#define Ee 256
#define Vv 80
#define HEADS 8
#define DHh 32
#define HID 256
#define BLANK 79
#define MASK_TOK 78
#define THRESH 0.7f

typedef __attribute__((ext_vector_type(16))) __bf16 v16bf;
typedef __attribute__((ext_vector_type(8)))  __bf16 v8bf;
typedef __attribute__((ext_vector_type(8)))  float  v8f;

__device__ __forceinline__ v8f wmma_bf16(v16bf a, v16bf b, v8f c) {
  return __builtin_amdgcn_wmma_f32_16x16x32_bf16(false, a, false, b, (short)0, c, false, false);
}

// ---- bf16 fragment loaders (A 16x32 / B 32x16, contiguous k = 16*(lane/16) + s) -----------
__device__ __forceinline__ v16bf load_a_bf(const __bf16* A, int lda, int m0, int k0, int lane) {
  int row = m0 + (lane & 15);
  int kh  = (lane >> 4) * 16;
  const __bf16* p = A + (size_t)row * lda + k0 + kh;
  v8bf lo = *(const v8bf*)p;          // 16B load
  v8bf hi = *(const v8bf*)(p + 8);    // 16B load
  return __builtin_shufflevector(lo, hi, 0,1,2,3,4,5,6,7,8,9,10,11,12,13,14,15);
}

__device__ __forceinline__ v16bf load_b_bf(const __bf16* W, int ldw, int n0, int k0, int lane) {
  int col = n0 + (lane & 15);
  int kh  = (lane >> 4) * 16;
  const __bf16* p = W + (size_t)col * ldw + k0 + kh;
  v8bf lo = *(const v8bf*)p;
  v8bf hi = *(const v8bf*)(p + 8);
  return __builtin_shufflevector(lo, hi, 0,1,2,3,4,5,6,7,8,9,10,11,12,13,14,15);
}

// B with element (k,n) at Bm[k*ldk + n] (K-major, strided) — for attention V.
__device__ __forceinline__ v16bf load_b_kn_bf(const __bf16* Bm, int ldk, int n0, int k0, int lane) {
  int col = n0 + (lane & 15);
  int kh  = (lane >> 4) * 16;
  const __bf16* base = Bm + (size_t)(k0 + kh) * ldk + col;
  v16bf b;
#pragma unroll
  for (int i = 0; i < 8; ++i) {
    b[2 * i]     = base[(size_t)(2 * i) * ldk];
    b[2 * i + 1] = base[(size_t)(2 * i + 1) * ldk];
  }
  return b;
}

__device__ __forceinline__ v8f load_c_rm(const float* C, int ldc, int m0, int n0, int lane) {
  v8f c;
  int n  = n0 + (lane & 15);
  int mb = m0 + (lane >> 4) * 8;
#pragma unroll
  for (int v = 0; v < 8; ++v) c[v] = C[(size_t)(mb + v) * ldc + n];
  return c;
}

__device__ __forceinline__ void store_c_f32(float* C, int ldc, int m0, int n0, v8f c,
                                            const float* bias, int lane) {
  int n  = n0 + (lane & 15);
  float bv = bias ? bias[n] : 0.0f;
  int mb = m0 + (lane >> 4) * 8;
#pragma unroll
  for (int v = 0; v < 8; ++v) C[(size_t)(mb + v) * ldc + n] = c[v] + bv;
}

__device__ __forceinline__ void store_c_bf(__bf16* C, int ldc, int m0, int n0, v8f c,
                                           const float* bias, int lane) {
  int n  = n0 + (lane & 15);
  float bv = bias ? bias[n] : 0.0f;
  int mb = m0 + (lane >> 4) * 8;
#pragma unroll
  for (int v = 0; v < 8; ++v) C[(size_t)(mb + v) * ldc + n] = (__bf16)(c[v] + bv);
}

// ---- fp32 -> bf16 bulk convert ----------------------------------------------
__global__ void cvt_bf16_kernel(const float* __restrict__ src, __bf16* __restrict__ dst, int n) {
  int i = (blockIdx.x * blockDim.x + threadIdx.x) * 8;
  if (i >= n) return;
  v8bf o;
#pragma unroll
  for (int j = 0; j < 8; ++j) o[j] = (__bf16)src[i + j];
  *(v8bf*)(dst + i) = o;
}

// ---- GEMM: C(MxN) = A(MxK,bf16) @ W^T(bf16) + bias --------------------------
// NT = n-tiles per wave (compile-time): wave computes a 16 x (16*NT) stripe,
// A fragment reused NT times, all acc[] indices constant -> no movrel.
template <int NT>
__global__ void gemm_bf16_kernel(const __bf16* __restrict__ A, int lda,
                                 const __bf16* __restrict__ W, int ldw,
                                 const float* __restrict__ bias,
                                 float* __restrict__ Cf, __bf16* __restrict__ Cbf, int ldc,
                                 int M, int N, int K) {
  int lane = threadIdx.x & 31;
  int wave = threadIdx.x >> 5;
  int ncb = N / (16 * NT);
  int nblk = (M >> 4) * ncb;
  int blk = blockIdx.x * 8 + wave;
  if (blk >= nblk) return;                       // wave-uniform
  int m0 = (blk / ncb) << 4;
  int n0 = (blk % ncb) * (16 * NT);
  v8f zero = {};
  v8f acc[NT];
#pragma unroll
  for (int j = 0; j < NT; ++j) acc[j] = zero;
  for (int k0 = 0; k0 < K; k0 += 32) {
    // speculative stream-prefetch of next weight block (safe past end: dropped)
    __builtin_prefetch(W + (size_t)(n0 + (lane & 15)) * ldw + k0 + 32, 0, 1);
    v16bf a = load_a_bf(A, lda, m0, k0, lane);   // 2 x b128
#pragma unroll
    for (int j = 0; j < NT; ++j) {
      v16bf b = load_b_bf(W, ldw, n0 + 16 * j, k0, lane);
      acc[j] = wmma_bf16(a, b, acc[j]);
    }
  }
#pragma unroll
  for (int j = 0; j < NT; ++j) {
    if (Cf)  store_c_f32(Cf,  ldc, m0, n0 + 16 * j, acc[j], bias, lane);
    if (Cbf) store_c_bf(Cbf, ldc, m0, n0 + 16 * j, acc[j], bias, lane);
  }
}

// ---- CTC stats --------------------------------------------------------------
__global__ void ctc_stats_kernel(const float* __restrict__ y,
                                 int* __restrict__ ids, float* __restrict__ probs) {
  int idx = blockIdx.x * blockDim.x + threadIdx.x;
  if (idx >= Tt * Bb) return;
  const float* row = y + (size_t)idx * Vv;
  float mx = row[0]; int am = 0;
  for (int v = 1; v < Vv; ++v) { float xv = row[v]; if (xv > mx) { mx = xv; am = v; } }
  float s = 0.f;
  for (int v = 0; v < Vv; ++v) s += expf(row[v] - mx);
  ids[idx] = am;
  probs[idx] = 1.0f / s;
}

// ---- CTC collapse -----------------------------------------------------------
__global__ void ctc_collapse_kernel(const int* __restrict__ ids, const float* __restrict__ probs,
                                    int* __restrict__ out_ids, float* __restrict__ masked_out) {
  int b = threadIdx.x;
  if (b >= Bb) return;
  for (int t = 0; t < Tt; ++t) { out_ids[t * Bb + b] = BLANK; masked_out[t * Bb + b] = 0.0f; }
  int rank = 0;
  int runval = ids[b];
  float runp = probs[b];
  for (int t = 1; t <= Tt; ++t) {
    int id = (t < Tt) ? ids[t * Bb + b] : -1;
    if (t == Tt || id != runval) {
      if (runval != BLANK) {
        bool low = runp < THRESH;
        out_ids[rank * Bb + b] = low ? MASK_TOK : runval;
        if (low) masked_out[rank * Bb + b] = 1.0f;
        ++rank;
      }
      runval = id;
      runp = (t < Tt) ? probs[t * Bb + b] : 0.0f;
    } else {
      runp = fmaxf(runp, probs[t * Bb + b]);
    }
  }
}

// ---- attention: one wave per (b, h, 16 query rows), bf16 QKV ----------------
__global__ void attn_kernel(const __bf16* __restrict__ QKV, __bf16* __restrict__ O) {
  __shared__ __align__(16) float  sc[16 * Tt];     // 32 KB fp32 scores
  __shared__ __align__(16) __bf16 pbf[16 * Tt];    // 16 KB bf16 probabilities
  int lane = threadIdx.x;
  int rb = blockIdx.x & 31;
  int bh = blockIdx.x >> 5;
  int b = bh >> 3, h = bh & 7;
  const int ldt = Bb * 768;
  const __bf16* Qb = QKV + (size_t)b * 768 + h * DHh;
  const __bf16* Kb = QKV + (size_t)b * 768 + Ee + h * DHh;
  const __bf16* Vb = QKV + (size_t)b * 768 + 2 * Ee + h * DHh;
  const float scale = 0.1767766952966369f;         // 1/sqrt(DH)

  v16bf aq = load_a_bf(Qb, ldt, rb * 16, 0, lane); // hoisted across all 32 s-tiles
  for (int st = 0; st < Tt / 16; ++st) {
    v16bf kb = load_b_bf(Kb, ldt, st * 16, 0, lane);
    v8f acc = {};
    acc = wmma_bf16(aq, kb, acc);
#pragma unroll
    for (int v = 0; v < 8; ++v) acc[v] *= scale;
    store_c_f32(sc, Tt, 0, st * 16, acc, nullptr, lane);
  }
  __syncthreads();
  if (lane < 16) {
    float* row = sc + lane * Tt;
    __bf16* prow = pbf + lane * Tt;
    float mx = -INFINITY;
    for (int s = 0; s < Tt; ++s) mx = fmaxf(mx, row[s]);
    float sum = 0.f;
    for (int s = 0; s < Tt; ++s) { float e = expf(row[s] - mx); row[s] = e; sum += e; }
    float inv = 1.0f / sum;
    for (int s = 0; s < Tt; ++s) prow[s] = (__bf16)(row[s] * inv);
  }
  __syncthreads();
  v8f acc0 = {}, acc1 = {};
  for (int k0 = 0; k0 < Tt; k0 += 32) {            // O = P(16xT) @ V(TxDH)
    v16bf a = load_a_bf(pbf, Tt, 0, k0, lane);     // 2 x ds_load_b128
    acc0 = wmma_bf16(a, load_b_kn_bf(Vb, ldt, 0,  k0, lane), acc0);
    acc1 = wmma_bf16(a, load_b_kn_bf(Vb, ldt, 16, k0, lane), acc1);
  }
  __bf16* Ob = O + (size_t)b * Ee + h * DHh;
  store_c_bf(Ob, Bb * Ee, rb * 16, 0,  acc0, nullptr, lane);
  store_c_bf(Ob, Bb * Ee, rb * 16, 16, acc1, nullptr, lane);
}

// ---- embedding gather into z_bf[:, :, 0:256] --------------------------------
__global__ void gather_kernel(const int* __restrict__ out_ids, const float* __restrict__ emb,
                              __bf16* __restrict__ z) {
  int tb = blockIdx.x;
  int e  = threadIdx.x;
  int id = out_ids[tb];
  z[(size_t)tb * 512 + e] = (__bf16)emb[id * Ee + e];
}

// ---- LSTM recurrence: block 0 = fwd, block 1 = bwd --------------------------
__global__ void lstm_scan_kernel(const float* __restrict__ Gf, const float* __restrict__ Gb,
                                 const __bf16* __restrict__ whh_f, const __bf16* __restrict__ whh_b,
                                 __bf16* __restrict__ hout) {
  extern __shared__ float smem[];
  float*  gbuf = smem;                               // 32 x 1024 fp32 (128 KB)
  float*  cbuf = smem + Bb * 1024;                   // 32 x 256  fp32 (32 KB)
  __bf16* hbf  = (__bf16*)(cbuf + Bb * HID);         // 32 x 256  bf16 (16 KB)
  int dir = blockIdx.x;
  const float*  G   = dir ? Gb : Gf;
  const __bf16* whh = dir ? whh_b : whh_f;
  int lane = threadIdx.x & 31;
  int wave = threadIdx.x >> 5;

  for (int i = threadIdx.x; i < Bb * HID; i += blockDim.x) {
    cbuf[i] = 0.f;
    hbf[i] = (__bf16)0.0f;
  }
  __syncthreads();

  int mt = wave & 1;
  int g0 = wave >> 1;                                // 0..3
  for (int step = 0; step < Tt; ++step) {
    int t = dir ? (Tt - 1 - step) : step;
    const float* Gt = G + (size_t)t * Bb * 1024;
    for (int ng = g0; ng < 16; ng += 4) {            // groups of 4 n-tiles
      int nt0 = ng << 2;
      v8f acc[4];
#pragma unroll
      for (int j = 0; j < 4; ++j)
        acc[j] = load_c_rm(Gt, 1024, mt * 16, (nt0 + j) * 16, lane);
#pragma unroll
      for (int k0 = 0; k0 < HID; k0 += 32) {
        v16bf a = load_a_bf(hbf, HID, mt * 16, k0, lane);  // ds b128 x2, reused x4
#pragma unroll
        for (int j = 0; j < 4; ++j) {
          v16bf b = load_b_bf(whh, HID, (nt0 + j) * 16, k0, lane);
          acc[j] = wmma_bf16(a, b, acc[j]);
        }
      }
#pragma unroll
      for (int j = 0; j < 4; ++j)
        store_c_f32(gbuf, 1024, mt * 16, (nt0 + j) * 16, acc[j], nullptr, lane);
    }
    __syncthreads();
    for (int i = threadIdx.x; i < Bb * HID; i += blockDim.x) {
      int b  = i >> 8;
      int hd = i & 255;
      float gi = gbuf[b * 1024 + hd];
      float gf = gbuf[b * 1024 + 256 + hd];
      float gg = gbuf[b * 1024 + 512 + hd];
      float go = gbuf[b * 1024 + 768 + hd];
      float c = (1.f / (1.f + expf(-gf))) * cbuf[i] + (1.f / (1.f + expf(-gi))) * tanhf(gg);
      cbuf[i] = c;
      float hv = (1.f / (1.f + expf(-go))) * tanhf(c);
      hbf[i] = (__bf16)hv;
      hout[((size_t)t * Bb + b) * 512 + dir * HID + hd] = (__bf16)hv;
    }
    __syncthreads();
  }
}

extern "C" void kernel_launch(void* const* d_in, const int* in_sizes, int n_in,
                              void* d_out, int out_size, void* d_ws, size_t ws_size,
                              hipStream_t stream) {
  const float* x          = (const float*)d_in[0];
  const float* y          = (const float*)d_in[1];
  const float* emb        = (const float*)d_in[2];
  const float* attn_in_w  = (const float*)d_in[3];
  const float* attn_in_b  = (const float*)d_in[4];
  const float* attn_out_w = (const float*)d_in[5];
  const float* attn_out_b = (const float*)d_in[6];
  const float* lin_w      = (const float*)d_in[7];
  const float* lin_b      = (const float*)d_in[8];
  const float* wih_l0f = (const float*)d_in[9];
  const float* whh_l0f = (const float*)d_in[10];
  const float* b_l0f   = (const float*)d_in[11];
  const float* wih_l0b = (const float*)d_in[12];
  const float* whh_l0b = (const float*)d_in[13];
  const float* b_l0b   = (const float*)d_in[14];
  const float* wih_l1f = (const float*)d_in[15];
  const float* whh_l1f = (const float*)d_in[16];
  const float* b_l1f   = (const float*)d_in[17];
  const float* wih_l1b = (const float*)d_in[18];
  const float* whh_l1b = (const float*)d_in[19];
  const float* b_l1b   = (const float*)d_in[20];
  (void)in_sizes; (void)n_in; (void)out_size; (void)ws_size;

  const int TB = Tt * Bb;                          // 16384
  char* ws = (char*)d_ws;
  size_t off = 0;
  auto carve = [&](size_t bytes) { void* p = ws + off; off += (bytes + 255) & ~(size_t)255; return p; };
  int*    ids     = (int*)   carve((size_t)TB * 4);
  float*  probs   = (float*) carve((size_t)TB * 4);
  int*    out_ids = (int*)   carve((size_t)TB * 4);
  __bf16* x_bf    = (__bf16*)carve((size_t)TB * 256 * 2);
  __bf16* QKV_bf  = (__bf16*)carve((size_t)TB * 768 * 2);
  __bf16* aO_bf   = (__bf16*)carve((size_t)TB * 256 * 2);
  __bf16* z_bf    = (__bf16*)carve((size_t)TB * 512 * 2);
  __bf16* h0_bf   = (__bf16*)carve((size_t)TB * 512 * 2);
  __bf16* h1_bf   = (__bf16*)carve((size_t)TB * 512 * 2);
  float*  Gfb     = (float*) carve((size_t)TB * 1024 * 4);
  float*  Gbb     = (float*) carve((size_t)TB * 1024 * 4);
  __bf16* inw_bf  = (__bf16*)carve((size_t)768 * 256 * 2);
  __bf16* outw_bf = (__bf16*)carve((size_t)256 * 256 * 2);
  __bf16* wih_bf[4]; __bf16* whh_bf[4];
  for (int i = 0; i < 4; ++i) {
    wih_bf[i] = (__bf16*)carve((size_t)1024 * 512 * 2);
    whh_bf[i] = (__bf16*)carve((size_t)1024 * 256 * 2);
  }
  __bf16* lin_bf  = (__bf16*)carve((size_t)Vv * 512 * 2);

  float* logits = (float*)d_out;
  float* masked = logits + (size_t)TB * Vv;

  auto cvt = [&](const float* s, __bf16* d, int n) {
    cvt_bf16_kernel<<<(n / 8 + 255) / 256, 256, 0, stream>>>(s, d, n);
  };
  auto gblocks = [](int M, int N, int NT) { int nb = (M >> 4) * (N / (16 * NT)); return (nb + 7) / 8; };
  const size_t LSTM_SMEM = (size_t)Bb * 1024 * 4 + (size_t)Bb * HID * 4 + (size_t)Bb * HID * 2;

  // 0) one-time weight / input conversions
  cvt(x, x_bf, TB * 256);
  cvt(attn_in_w, inw_bf, 768 * 256);
  cvt(attn_out_w, outw_bf, 256 * 256);
  const float* wihs[4] = {wih_l0f, wih_l0b, wih_l1f, wih_l1b};
  const float* whhs[4] = {whh_l0f, whh_l0b, whh_l1f, whh_l1b};
  for (int i = 0; i < 4; ++i) {
    cvt(wihs[i], wih_bf[i], 1024 * 512);
    cvt(whhs[i], whh_bf[i], 1024 * 256);
  }
  cvt(lin_w, lin_bf, Vv * 512);

  // 1) CTC decode
  ctc_stats_kernel<<<TB / 256, 256, 0, stream>>>(y, ids, probs);
  ctc_collapse_kernel<<<1, 32, 0, stream>>>(ids, probs, out_ids, masked);

  // 2) MHA
  gemm_bf16_kernel<4><<<gblocks(TB, 768, 4), 256, 0, stream>>>(
      x_bf, 256, inw_bf, 256, attn_in_b, nullptr, QKV_bf, 768, TB, 768, 256);
  attn_kernel<<<Bb * HEADS * (Tt / 16), 32, 0, stream>>>(QKV_bf, aO_bf);
  gemm_bf16_kernel<4><<<gblocks(TB, 256, 4), 256, 0, stream>>>(
      aO_bf, 256, outw_bf, 256, attn_out_b, nullptr, z_bf + 256, 512, TB, 256, 256);
  // 3) z[:, :, 0:256] = emb[out_ids]
  gather_kernel<<<TB, 256, 0, stream>>>(out_ids, emb, z_bf);

  // 4) BiLSTM layer 0
  gemm_bf16_kernel<4><<<gblocks(TB, 1024, 4), 256, 0, stream>>>(
      z_bf, 512, wih_bf[0], 512, b_l0f, Gfb, nullptr, 1024, TB, 1024, 512);
  gemm_bf16_kernel<4><<<gblocks(TB, 1024, 4), 256, 0, stream>>>(
      z_bf, 512, wih_bf[1], 512, b_l0b, Gbb, nullptr, 1024, TB, 1024, 512);
  lstm_scan_kernel<<<2, 256, LSTM_SMEM, stream>>>(Gfb, Gbb, whh_bf[0], whh_bf[1], h0_bf);

  // 5) BiLSTM layer 1
  gemm_bf16_kernel<4><<<gblocks(TB, 1024, 4), 256, 0, stream>>>(
      h0_bf, 512, wih_bf[2], 512, b_l1f, Gfb, nullptr, 1024, TB, 1024, 512);
  gemm_bf16_kernel<4><<<gblocks(TB, 1024, 4), 256, 0, stream>>>(
      h0_bf, 512, wih_bf[3], 512, b_l1b, Gbb, nullptr, 1024, TB, 1024, 512);
  lstm_scan_kernel<<<2, 256, LSTM_SMEM, stream>>>(Gfb, Gbb, whh_bf[2], whh_bf[3], h1_bf);

  // 6) logits: N = 80 -> one 16x80 stripe per wave (NT=5)
  gemm_bf16_kernel<5><<<gblocks(TB, Vv, 5), 256, 0, stream>>>(
      h1_bf, 512, lin_bf, 512, lin_b, logits, nullptr, Vv, TB, Vv, 512);
}
